// GlobalAttention_7868380086506
// MI455X (gfx1250) — matrix operand order repaired and checked by reading, not compiled, not run on hardware
//
#include <hip/hip_runtime.h>
#include <hip/hip_bf16.h>

#define BB 2
#define NN 2048
#define CC 1024
#define HH 16
#define HD 64
#define C3 3072

typedef __attribute__((ext_vector_type(16))) __bf16 v16bf;
typedef __attribute__((ext_vector_type(8)))  __bf16 v8bf;
typedef __attribute__((ext_vector_type(4)))  __bf16 v4bf;
typedef __attribute__((ext_vector_type(8)))  float  v8f;
typedef __attribute__((ext_vector_type(4)))  float  v4f;

static __device__ __forceinline__ __bf16 f2bf(float f) { return (__bf16)f; }
static __device__ __forceinline__ float bf2f(__bf16 b) { return (float)b; }

static __device__ __forceinline__ v16bf cat8(v8bf a, v8bf b) {
  return __builtin_shufflevector(a, b, 0, 1, 2, 3, 4, 5, 6, 7,
                                 8, 9, 10, 11, 12, 13, 14, 15);
}

static __device__ __forceinline__ v8f wmma_bf16(v16bf a, v16bf b, v8f c) {
  return __builtin_amdgcn_wmma_f32_16x16x32_bf16(false, a, false, b, (short)0, c,
                                                 false, false);
}

// ---------------------------------------------------------------------------
// Prep kernel A: x (fp32 [M][C]) -> bf16, same layout. 4 elems/thread.
// ---------------------------------------------------------------------------
__global__ __launch_bounds__(256)
void cvt_x_kernel(const float* __restrict__ x, __bf16* __restrict__ xb) {
  const size_t i = ((size_t)blockIdx.x * blockDim.x + threadIdx.x) * 4;
  v4f f = *(const v4f*)(x + i);
  v4bf o;
  o[0] = f2bf(f[0]); o[1] = f2bf(f[1]); o[2] = f2bf(f[2]); o[3] = f2bf(f[3]);
  *(v4bf*)(xb + i) = o;
}

// ---------------------------------------------------------------------------
// Prep kernel B: W (fp32 [K][N]) -> Wt (bf16 [N][K]).  Coalesced writes.
// ---------------------------------------------------------------------------
__global__ __launch_bounds__(256)
void cvt_wt_kernel(const float* __restrict__ W, __bf16* __restrict__ Wt,
                   int K, int Ncols) {
  const size_t i = (size_t)blockIdx.x * blockDim.x + threadIdx.x; // over N*K
  const int k = (int)(i & (size_t)(K - 1));   // K == 1024
  const int n = (int)(i >> 10);
  Wt[i] = f2bf(W[(size_t)k * Ncols + n]);
}

// ---------------------------------------------------------------------------
// Kernel 1: qkv = x @ Wqkv + bqkv   (M=4096, K=1024, N=3072), bf16 WMMA.
// 2x2 register blocking: each wave owns a 32x32 output block (4 accumulators),
// reusing each A fragment across 2 N-tiles and each B fragment across 2 M-tiles.
// Epilogue scatters q,k -> [B,H,N,HD] and v -> [B,H,HD,N] in ws.
// ---------------------------------------------------------------------------
__global__ __launch_bounds__(128)
void qkv_gemm_kernel(const __bf16* __restrict__ xb, const __bf16* __restrict__ Wt,
                     const float* __restrict__ bias,
                     __bf16* __restrict__ qbuf, __bf16* __restrict__ kbuf,
                     __bf16* __restrict__ vt) {
  const int lane = threadIdx.x & 31;
  const int wave = threadIdx.x >> 5;
  const int half = lane >> 4;
  const int l16  = lane & 15;
  const int mt = blockIdx.y;                 // 0..127  (32-row blocks)
  const int nt = blockIdx.x * 4 + wave;      // 0..95   (32-col blocks)

  const __bf16* arow0 = xb + (size_t)(mt * 32 + l16) * CC;
  const __bf16* arow1 = arow0 + (size_t)16 * CC;
  const __bf16* wrow0 = Wt + (size_t)(nt * 32 + l16) * CC;
  const __bf16* wrow1 = wrow0 + (size_t)16 * CC;

  v8f acc00 = {}, acc01 = {}, acc10 = {}, acc11 = {};
  for (int kk = 0; kk < CC; kk += 32) {
    const int ka = kk + half * 16;           // A: K contiguous 16 per half
    const int kb = kk + half * 16;           // B: K contiguous 16 per half
    v16bf a0 = cat8(*(const v8bf*)(arow0 + ka), *(const v8bf*)(arow0 + ka + 8));
    v16bf a1 = cat8(*(const v8bf*)(arow1 + ka), *(const v8bf*)(arow1 + ka + 8));
    v16bf b0 = *(const v16bf*)(wrow0 + kb);
    v16bf b1 = *(const v16bf*)(wrow1 + kb);
    __builtin_prefetch(wrow0 + kk + 128, 0, 1);
    __builtin_prefetch(wrow1 + kk + 128, 0, 1);
    acc00 = wmma_bf16(a0, b0, acc00);
    acc01 = wmma_bf16(a0, b1, acc01);
    acc10 = wmma_bf16(a1, b0, acc10);
    acc11 = wmma_bf16(a1, b1, acc11);
  }

#pragma unroll
  for (int ns = 0; ns < 2; ++ns) {
    const int n   = nt * 32 + ns * 16 + l16;
    const float bv = bias[n];
    const int s   = n >> 10;                 // 0=q,1=k,2=v
    const int rem = n & (CC - 1);
    const int h   = rem >> 6;
    const int d   = rem & (HD - 1);
#pragma unroll
    for (int ms = 0; ms < 2; ++ms) {
      const v8f acc = (ns == 0) ? (ms == 0 ? acc00 : acc10)
                                : (ms == 0 ? acc01 : acc11);
#pragma unroll
      for (int i = 0; i < 8; ++i) {
        const int row = mt * 32 + ms * 16 + i + half * 8;
        const int b   = row >> 11;
        const int tok = row & (NN - 1);
        const __bf16 val = f2bf(acc[i] + bv);
        if (s == 0)
          qbuf[(((size_t)(b * HH + h)) * NN + tok) * HD + d] = val;
        else if (s == 1)
          kbuf[(((size_t)(b * HH + h)) * NN + tok) * HD + d] = val;
        else
          vt[(((size_t)(b * HH + h)) * HD + d) * NN + tok] = val;
      }
    }
  }
}

// ---------------------------------------------------------------------------
// Kernel 2: in-place S2-RoPE on q,k where rope_mask set. One thread per pair.
// ---------------------------------------------------------------------------
__global__ __launch_bounds__(256)
void rope_kernel(__bf16* __restrict__ qbuf, __bf16* __restrict__ kbuf,
                 const float* __restrict__ phx, const float* __restrict__ phy,
                 const float* __restrict__ phs,
                 const unsigned char* __restrict__ mask) {
  const int idx = blockIdx.x * blockDim.x + threadIdx.x;   // B*H*N*32 threads
  const int p   = idx & 31;
  const int t   = idx >> 5;
  const int tok = t & (NN - 1);
  const int t2  = t >> 11;
  const int h   = t2 & (HH - 1);
  const int b   = t2 >> 4;
  if (b >= BB) return;
  if (!mask[b * NN + tok]) return;

  int d0, d1; float ph;
  if (p < 12)      { d0 = p;             d1 = d0 + 12; ph = phx[(b * NN + tok) * 12 + p]; }
  else if (p < 24) { d0 = 24 + (p - 12); d1 = d0 + 12; ph = phy[(b * NN + tok) * 12 + (p - 12)]; }
  else             { d0 = 48 + (p - 24); d1 = d0 + 8;  ph = phs[(b * NN + tok) * 8 + (p - 24)]; }

  float c, sn;
  __sincosf(ph, &sn, &c);
  const size_t base = (((size_t)(b * HH + h)) * NN + tok) * HD;

  float q0 = bf2f(qbuf[base + d0]), q1 = bf2f(qbuf[base + d1]);
  qbuf[base + d0] = f2bf(q0 * c - q1 * sn);
  qbuf[base + d1] = f2bf(q1 * c + q0 * sn);
  float k0 = bf2f(kbuf[base + d0]), k1 = bf2f(kbuf[base + d1]);
  kbuf[base + d0] = f2bf(k0 * c - k1 * sn);
  kbuf[base + d1] = f2bf(k1 * c + k0 * sn);
}

// ---------------------------------------------------------------------------
// Kernel 3: flash attention, one wave per (b,h,16-query tile).
// S^T = K_tile @ Q^T so exp(S) A-fragment for P@V is built in-lane.
// ---------------------------------------------------------------------------
__global__ __launch_bounds__(128)
void attn_kernel(const __bf16* __restrict__ qbuf, const __bf16* __restrict__ kbuf,
                 const __bf16* __restrict__ vt, __bf16* __restrict__ aout) {
  const int lane = threadIdx.x & 31;
  const int wave = threadIdx.x >> 5;
  const int half = lane >> 4;
  const int l16  = lane & 15;
  const int bh = blockIdx.y;                 // 0..31
  const int b  = bh >> 4;
  const int h  = bh & (HH - 1);
  const int qt = blockIdx.x * 4 + wave;      // 0..127

  const __bf16* qbase = qbuf + ((size_t)bh) * NN * HD;
  const __bf16* kbase = kbuf + ((size_t)bh) * NN * HD;
  const __bf16* vbase = vt   + ((size_t)bh) * HD * NN;

  // Q as B-operand: lane = query col, K-dim (d) split {0..7,16..23}/{8..15,24..31}.
  const __bf16* qrow = qbase + (size_t)(qt * 16 + l16) * HD;
  const v16bf qb0 = cat8(*(const v8bf*)(qrow + half * 8), *(const v8bf*)(qrow + 16 + half * 8));        // d 0..31
  const v16bf qb1 = cat8(*(const v8bf*)(qrow + 32 + half * 8), *(const v8bf*)(qrow + 48 + half * 8));   // d 32..63

  v8f o[4]; o[0] = v8f{}; o[1] = v8f{}; o[2] = v8f{}; o[3] = v8f{};
  float mrun = -1e30f, lrun = 0.f;
  const int da = half * 8;                   // A-fragment d offset

  for (int kt = 0; kt < NN; kt += 32) {
    const __bf16* kr0 = kbase + (size_t)(kt + l16) * HD;
    const __bf16* kr1 = kbase + (size_t)(kt + 16 + l16) * HD;
    v16bf ka0a = cat8(*(const v8bf*)(kr0 + da), *(const v8bf*)(kr0 + da + 16));
    v16bf ka0b = cat8(*(const v8bf*)(kr0 + 32 + da), *(const v8bf*)(kr0 + 32 + da + 16));
    v16bf ka1a = cat8(*(const v8bf*)(kr1 + da), *(const v8bf*)(kr1 + da + 16));
    v16bf ka1b = cat8(*(const v8bf*)(kr1 + 32 + da), *(const v8bf*)(kr1 + 32 + da + 16));

    v8f st0 = {}; st0 = wmma_bf16(ka0a, qb0, st0); st0 = wmma_bf16(ka0b, qb1, st0);
    v8f st1 = {}; st1 = wmma_bf16(ka1a, qb0, st1); st1 = wmma_bf16(ka1b, qb1, st1);

    // Online softmax: per lane query = l16; in-lane keys + partner half.
    float cmax = -1e30f;
#pragma unroll
    for (int j = 0; j < 8; ++j) {
      st0[j] *= 0.125f; st1[j] *= 0.125f;   // 1/sqrt(HD)
      cmax = fmaxf(cmax, fmaxf(st0[j], st1[j]));
    }
    cmax = fmaxf(cmax, __shfl_xor(cmax, 16, 32));
    const float mnew = fmaxf(mrun, cmax);
    const float corr = __expf(mrun - mnew);
    float ssum = 0.f;
    v16bf pa;
#pragma unroll
    for (int j = 0; j < 8; ++j) {
      float p0 = __expf(st0[j] - mnew);
      float p1 = __expf(st1[j] - mnew);
      ssum += p0 + p1;
      pa[j] = f2bf(p0); pa[8 + j] = f2bf(p1);
    }
    ssum += __shfl_xor(ssum, 16, 32);
    lrun = lrun * corr + ssum;
    mrun = mnew;

    // Rescale O: lane-indexed corr -> VGPR-row-indexed factors.
#pragma unroll
    for (int j = 0; j < 8; ++j) {
      const float cj = __shfl(corr, j + (half << 3), 32);
      o[0][j] *= cj; o[1][j] *= cj; o[2][j] *= cj; o[3][j] *= cj;
    }

    // P @ V: V^T stored [hd][n] -> B-operand keys split {0..7,16..23}/{8..15,24..31}.
#pragma unroll
    for (int t = 0; t < 4; ++t) {
      const v16bf vb =
          cat8(*(const v8bf*)(vbase + (size_t)(t * 16 + l16) * NN + kt + half * 8), *(const v8bf*)(vbase + (size_t)(t * 16 + l16) * NN + kt + 16 + half * 8));
      o[t] = wmma_bf16(pa, vb, o[t]);
    }
  }

  // Final 1/l normalization, store [B,N,C] bf16.
  const float inv = 1.0f / lrun;
#pragma unroll
  for (int j = 0; j < 8; ++j) {
    const float rj = __shfl(inv, j + (half << 3), 32);
    const int row = qt * 16 + j + half * 8;
    const size_t off = ((size_t)b * NN + row) * CC + h * HD;
#pragma unroll
    for (int t = 0; t < 4; ++t)
      aout[off + t * 16 + l16] = f2bf(o[t][j] * rj);
  }
}

// ---------------------------------------------------------------------------
// Kernel 4: out = attn @ Wproj + bproj  (4096 x 1024 x 1024), fp32 out.
// 2x2 register blocking, same scheme as kernel 1.
// ---------------------------------------------------------------------------
__global__ __launch_bounds__(128)
void proj_gemm_kernel(const __bf16* __restrict__ a, const __bf16* __restrict__ Wt,
                      const float* __restrict__ bias, float* __restrict__ out) {
  const int lane = threadIdx.x & 31;
  const int wave = threadIdx.x >> 5;
  const int half = lane >> 4;
  const int l16  = lane & 15;
  const int mt = blockIdx.y;                 // 0..127
  const int nt = blockIdx.x * 4 + wave;      // 0..31

  const __bf16* arow0 = a + (size_t)(mt * 32 + l16) * CC;
  const __bf16* arow1 = arow0 + (size_t)16 * CC;
  const __bf16* wrow0 = Wt + (size_t)(nt * 32 + l16) * CC;
  const __bf16* wrow1 = wrow0 + (size_t)16 * CC;

  v8f acc00 = {}, acc01 = {}, acc10 = {}, acc11 = {};
  for (int kk = 0; kk < CC; kk += 32) {
    const int ka = kk + half * 16;
    const int kb = kk + half * 16;
    v16bf a0 = cat8(*(const v8bf*)(arow0 + ka), *(const v8bf*)(arow0 + ka + 8));
    v16bf a1 = cat8(*(const v8bf*)(arow1 + ka), *(const v8bf*)(arow1 + ka + 8));
    v16bf b0 = *(const v16bf*)(wrow0 + kb);
    v16bf b1 = *(const v16bf*)(wrow1 + kb);
    __builtin_prefetch(wrow0 + kk + 128, 0, 1);
    __builtin_prefetch(wrow1 + kk + 128, 0, 1);
    acc00 = wmma_bf16(a0, b0, acc00);
    acc01 = wmma_bf16(a0, b1, acc01);
    acc10 = wmma_bf16(a1, b0, acc10);
    acc11 = wmma_bf16(a1, b1, acc11);
  }

#pragma unroll
  for (int ns = 0; ns < 2; ++ns) {
    const int n = nt * 32 + ns * 16 + l16;
    const float bv = bias[n];
#pragma unroll
    for (int ms = 0; ms < 2; ++ms) {
      const v8f acc = (ns == 0) ? (ms == 0 ? acc00 : acc10)
                                : (ms == 0 ? acc01 : acc11);
#pragma unroll
      for (int i = 0; i < 8; ++i) {
        const int row = mt * 32 + ms * 16 + i + half * 8;
        out[(size_t)row * CC + n] = acc[i] + bv;
      }
    }
  }
}

// ---------------------------------------------------------------------------
extern "C" void kernel_launch(void* const* d_in, const int* in_sizes, int n_in,
                              void* d_out, int out_size, void* d_ws, size_t ws_size,
                              hipStream_t stream) {
  const float* x    = (const float*)d_in[0];
  const float* phx  = (const float*)d_in[1];
  const float* phy  = (const float*)d_in[2];
  const float* phs  = (const float*)d_in[3];
  const unsigned char* mask = (const unsigned char*)d_in[4];
  const float* Wqkv = (const float*)d_in[5];
  const float* bqkv = (const float*)d_in[6];
  const float* Wpro = (const float*)d_in[7];
  const float* bpro = (const float*)d_in[8];
  float* outp = (float*)d_out;

  const size_t per = (size_t)BB * HH * NN * HD;   // 4 Mi elements
  __bf16* qbuf  = (__bf16*)d_ws;                  // 8 MB
  __bf16* kbuf  = qbuf + per;                     // 8 MB
  __bf16* vtb   = kbuf + per;                     // 8 MB
  __bf16* aout  = vtb + per;                      // 8 MB
  __bf16* xb    = aout + per;                     // 8 MB
  __bf16* wqkvt = xb + (size_t)BB * NN * CC;      // 6 MB
  __bf16* wprot = wqkvt + (size_t)C3 * CC;        // 2 MB   (48 MB total)

  // 0) One-shot bf16 conversion / weight transposition.
  cvt_x_kernel<<<(BB * NN * CC) / (256 * 4), 256, 0, stream>>>(x, xb);
  cvt_wt_kernel<<<(C3 * CC) / 256, 256, 0, stream>>>(Wqkv, wqkvt, CC, C3);
  cvt_wt_kernel<<<(CC * CC) / 256, 256, 0, stream>>>(Wpro, wprot, CC, CC);

  // 1) QKV GEMM + scatter (2x2 blocked waves: 32x32 per wave)
  qkv_gemm_kernel<<<dim3(24, 128), 128, 0, stream>>>(xb, wqkvt, bqkv,
                                                     qbuf, kbuf, vtb);
  // 2) RoPE on q,k
  rope_kernel<<<(BB * HH * NN * 32) / 256, 256, 0, stream>>>(qbuf, kbuf,
                                                             phx, phy, phs, mask);
  // 3) Attention
  attn_kernel<<<dim3(NN / 64, BB * HH), 128, 0, stream>>>(qbuf, kbuf, vtb, aout);
  // 4) Output projection (2x2 blocked)
  proj_gemm_kernel<<<dim3(8, 128), 128, 0, stream>>>(aout, wprot, bpro, outp);
}
